// PPNSemEnc_78443282694824
// MI455X (gfx1250) — matrix operand order repaired and checked by reading, not compiled, not run on hardware
//
#include <hip/hip_runtime.h>
#include <stdint.h>

// ---------------------------------------------------------------------------
// MI455X (gfx1250) implementation. wave32. Matrix work goes through
// v_wmma_f32_16x16x32_bf16 with a 2-term bf16 split (hi*hi + hi*lo + lo*hi)
// for ~f32 accuracy at bf16-WMMA throughput. Segment mean is a streaming
// pass over image_feats (the 819MB roofline term, ~35us at 23.3TB/s) with
// native agent-scope f32 atomics into an L2-resident 8MB accumulator.
// ---------------------------------------------------------------------------

typedef unsigned short u16;
typedef __attribute__((ext_vector_type(16))) __bf16 v16bf;
typedef __attribute__((ext_vector_type(8)))  float  v8f;
typedef __attribute__((ext_vector_type(4)))  unsigned int u32x4;

#define WMMA_BF16(a, b, c) \
  __builtin_amdgcn_wmma_f32_16x16x32_bf16(false, (a), false, (b), (short)0, (c), false, false)

#define COS_THRESH 0.7660444431189780f   /* cos(40 deg) */
#define T_SOFT     10.0f
#define NEG_RAW    (-9.0e16f)            /* NEG(-9e15) * T(10), as in reference */

__device__ __forceinline__ u16 f32_to_bf16(float f) {
  unsigned int u = __float_as_uint(f);
  u += 0x7fffu + ((u >> 16) & 1u);       // round-to-nearest-even
  return (u16)(u >> 16);
}
__device__ __forceinline__ float bf16_to_f32(u16 h) {
  return __uint_as_float(((unsigned int)h) << 16);
}
__device__ __forceinline__ void split2(float f, u16* hi, u16* lo) {
  u16 h = f32_to_bf16(f);
  *hi = h;
  *lo = f32_to_bf16(f - bf16_to_f32(h));
}

// Native hardware f32 atomic add (global_atomic_add_f32), no CAS loop:
// relaxed, device(agent)-scope FP atomics are supported by the CDNA5 L2.
__device__ __forceinline__ void atomAddF(float* p, float v) {
#ifdef __HIP_MEMORY_SCOPE_AGENT
  (void)__hip_atomic_fetch_add(p, v, __ATOMIC_RELAXED, __HIP_MEMORY_SCOPE_AGENT);
#else
  atomicAdd(p, v);
#endif
}

// --------------------------- utility kernels -------------------------------

__global__ void k_zero(float* __restrict__ p, long long n) {
  long long i = (long long)blockIdx.x * blockDim.x + threadIdx.x;
  long long st = (long long)gridDim.x * blockDim.x;
  for (; i < n; i += st) p[i] = 0.0f;
}

// One block per image row: stream 2048 f32 and atomically fold into sums[label].
__global__ __launch_bounds__(256)
void k_segsum(const float* __restrict__ img, const int* __restrict__ labels,
              float* __restrict__ sums, float* __restrict__ counts, int N, int DI) {
  int row = blockIdx.x;
  if (row >= N) return;
  int lab = labels[row];
  const float4* src = (const float4*)(img + (size_t)row * DI);
  float* dst = sums + (size_t)lab * DI;
  int nv = DI >> 2;
  for (int q = threadIdx.x; q < nv; q += blockDim.x) {
    float4 v = src[q];
    atomAddF(dst + 4 * q + 0, v.x);
    atomAddF(dst + 4 * q + 1, v.y);
    atomAddF(dst + 4 * q + 2, v.z);
    atomAddF(dst + 4 * q + 3, v.w);
  }
  if (threadIdx.x == 0) atomAddF(counts + lab, 1.0f);
}

// Per class: w = softmax(attr @ centroids^T), biasmix = w @ expert_b.
__global__ void k_mixw(const float* __restrict__ attr, const float* __restrict__ cent,
                       const float* __restrict__ eb, float* __restrict__ w,
                       float* __restrict__ biasmix, int C, int CP, int K, int D) {
  int c = blockIdx.x * blockDim.x + threadIdx.x;
  if (c >= CP) return;
  if (c >= C) {
    for (int k = 0; k < K; ++k) w[(size_t)c * K + k] = 0.f;
    for (int o = 0; o < D; ++o) biasmix[(size_t)c * D + o] = 0.f;
    return;
  }
  float s[16];
  for (int k = 0; k < K; ++k) s[k] = 0.f;
  const float* a = attr + (size_t)c * D;
  for (int d = 0; d < D; ++d) {
    float av = a[d];
    for (int k = 0; k < K; ++k) s[k] = fmaf(av, cent[(size_t)k * D + d], s[k]);
  }
  float mx = s[0];
  for (int k = 1; k < K; ++k) mx = fmaxf(mx, s[k]);
  float tot = 0.f;
  for (int k = 0; k < K; ++k) { s[k] = expf(s[k] - mx); tot += s[k]; }
  float inv = 1.f / tot;
  for (int k = 0; k < K; ++k) { s[k] *= inv; w[(size_t)c * K + k] = s[k]; }
  for (int o = 0; o < D; ++o) {
    float acc = 0.f;
    for (int k = 0; k < K; ++k) acc = fmaf(s[k], eb[(size_t)k * D + o], acc);
    biasmix[(size_t)c * D + o] = acc;
  }
}

// Z[c, k*D+d] = w[c,k] * attr[c,d]  (bf16 hi/lo), padded rows zero.
__global__ void k_buildZ(const float* __restrict__ attr, const float* __restrict__ w,
                         u16* __restrict__ zhi, u16* __restrict__ zlo,
                         int C, int CP, int K, int D) {
  long long KD = (long long)K * D;
  long long n = (long long)CP * KD;
  long long st = (long long)gridDim.x * blockDim.x;
  for (long long i = (long long)blockIdx.x * blockDim.x + threadIdx.x; i < n; i += st) {
    long long c = i / KD, j = i % KD;
    int k = (int)(j / D), d = (int)(j % D);
    float v = 0.f;
    if (c < C) v = w[c * K + k] * attr[c * D + d];
    split2(v, zhi + i, zlo + i);
  }
}

// W2t[o, k*D+d] = expert_W[k,d,o]  (pre-transposed B for contiguous frags).
__global__ void k_buildW2t(const float* __restrict__ eW, u16* __restrict__ hi,
                           u16* __restrict__ lo, int K, int D) {
  long long KD = (long long)K * D;
  long long n = (long long)D * KD;
  long long st = (long long)gridDim.x * blockDim.x;
  for (long long i = (long long)blockIdx.x * blockDim.x + threadIdx.x; i < n; i += st) {
    long long o = i / KD, j = i % KD;
    split2(eW[j * D + o], hi + i, lo + i);
  }
}

// attgT[hcol, d] = att_g[d, hcol]
__global__ void k_buildAttgT(const float* __restrict__ ag, u16* __restrict__ hi,
                             u16* __restrict__ lo, int D, int H) {
  long long n = (long long)D * H;
  long long st = (long long)gridDim.x * blockDim.x;
  for (long long i = (long long)blockIdx.x * blockDim.x + threadIdx.x; i < n; i += st) {
    long long hcol = i / D, d = i % D;
    split2(ag[d * H + hcol], hi + i, lo + i);
  }
}

// L2-normalize each row of g, emit bf16 hi/lo.
__global__ __launch_bounds__(256)
void k_rownorm(const float* __restrict__ g, u16* __restrict__ gnhi,
               u16* __restrict__ gnlo, int H) {
  __shared__ float red[256];
  int c = blockIdx.x;
  const float* row = g + (size_t)c * H;
  float ss = 0.f;
  for (int j = threadIdx.x; j < H; j += 256) { float v = row[j]; ss = fmaf(v, v, ss); }
  red[threadIdx.x] = ss; __syncthreads();
  for (int s = 128; s > 0; s >>= 1) {
    if (threadIdx.x < s) red[threadIdx.x] += red[threadIdx.x + s];
    __syncthreads();
  }
  float inv = 1.0f / (sqrtf(red[0]) + 1e-12f);
  for (int j = threadIdx.x; j < H; j += 256) {
    size_t idx = (size_t)c * H + j;
    split2(row[j] * inv, gnhi + idx, gnlo + idx);
  }
}

// raw = where(sim > thresh, sim, NEG) * T ; row softmax ; emit bf16 hi/lo.
__global__ __launch_bounds__(256)
void k_attn_softmax(const float* __restrict__ sim, u16* __restrict__ ahi,
                    u16* __restrict__ alo, int C, int CP) {
  __shared__ float red[256];
  int c = blockIdx.x;
  if (c >= C) {
    for (int j = threadIdx.x; j < CP; j += 256) {
      ahi[(size_t)c * CP + j] = 0; alo[(size_t)c * CP + j] = 0;
    }
    return;
  }
  const float* row = sim + (size_t)c * CP;
  float mx = -3.4e38f;
  for (int j = threadIdx.x; j < C; j += 256) {
    float s = row[j];
    float r = (s > COS_THRESH) ? s * T_SOFT : NEG_RAW;
    mx = fmaxf(mx, r);
  }
  red[threadIdx.x] = mx; __syncthreads();
  for (int s = 128; s > 0; s >>= 1) {
    if (threadIdx.x < s) red[threadIdx.x] = fmaxf(red[threadIdx.x], red[threadIdx.x + s]);
    __syncthreads();
  }
  mx = red[0]; __syncthreads();
  float tot = 0.f;
  for (int j = threadIdx.x; j < C; j += 256) {
    float s = row[j];
    float r = (s > COS_THRESH) ? s * T_SOFT : NEG_RAW;
    tot += expf(r - mx);
  }
  red[threadIdx.x] = tot; __syncthreads();
  for (int s = 128; s > 0; s >>= 1) {
    if (threadIdx.x < s) red[threadIdx.x] += red[threadIdx.x + s];
    __syncthreads();
  }
  float inv = 1.f / red[0];
  for (int j = threadIdx.x; j < CP; j += 256) {
    float v = 0.f;
    if (j < C) {
      float s = row[j];
      float r = (s > COS_THRESH) ? s * T_SOFT : NEG_RAW;
      v = expf(r - mx) * inv;
    }
    split2(v, ahi + (size_t)c * CP + j, alo + (size_t)c * CP + j);
  }
}

// protoT[j, c] = means[tpl[c], j]  (feature-major, pre-transposed B), pads zero.
__global__ void k_protoT(const float* __restrict__ sums, const float* __restrict__ counts,
                         const int* __restrict__ tpl, u16* __restrict__ phi,
                         u16* __restrict__ plo, int C, int CP, int DI) {
  long long n = (long long)DI * CP;
  long long st = (long long)gridDim.x * blockDim.x;
  for (long long i = (long long)blockIdx.x * blockDim.x + threadIdx.x; i < n; i += st) {
    int cc = (int)(i % CP);
    long long j = i / CP;
    float v = 0.f;
    if (cc < C) {
      int t = tpl[cc];
      v = sums[(size_t)t * DI + j] / fmaxf(counts[t], 1.0f);
    }
    split2(v, phi + i, plo + i);
  }
}

// ------------------------------ WMMA GEMM ----------------------------------
// A: (Mp x Kp) bf16 hi/lo, Bt: (Np x Kp) bf16 hi/lo (pre-transposed B).
// D = A*B + Cinit, f32 accum via 3x bf16 WMMA split. Each wave owns a 16x32
// output pair (two independent accumulators for ILP across WMMA hazards).
// Fragment mapping per CDNA5 ISA 16-bit A 16x32 layout:
//   lane l: row = l&15, h = l>>4; comps0-7 -> K=k0+8h.., comps8-15 -> K=k0+16+8h..
// Addressing uses uniform base + 32-bit per-lane element offsets (all operand
// buffers < 4GiB) so the backend can use SADDR-form global loads and keep the
// k-loop offset arithmetic 32-bit.
__device__ __forceinline__ v16bf load_frag(const u16* __restrict__ base, unsigned eo) {
  union { u32x4 q[2]; v16bf v; } f;
  f.q[0] = *(const u32x4*)(base + eo);
  f.q[1] = *(const u32x4*)(base + eo + 16);
  return f.v;
}

__global__ __launch_bounds__(256)
void k_gemm(const u16* __restrict__ Ahi, const u16* __restrict__ Alo,
            const u16* __restrict__ Bthi, const u16* __restrict__ Btlo,
            const float* __restrict__ Cinit, float* __restrict__ Dout,
            u16* __restrict__ Dhi, u16* __restrict__ Dlo,
            int Mp, int Np, int Kp, int Mreal) {
  const int tilesX = Np >> 5;                 // 32 output cols per wave
  const int ntiles = (Mp >> 4) * tilesX;
  const int wid = blockIdx.x * 8 + (int)(threadIdx.x >> 5);
  if (wid >= ntiles) return;
  const int m0 = (wid / tilesX) << 4;
  const int n0 = (wid % tilesX) << 5;
  const int lane = threadIdx.x & 31;
  const int lr = lane & 15;
  const int h  = lane >> 4;
  // 32-bit element offsets of each fragment's first 16B packet (h-dependent)
  unsigned aoff  = (unsigned)(m0 + lr) * (unsigned)Kp + (unsigned)(h << 3);
  unsigned boff0 = (unsigned)(n0 + lr) * (unsigned)Kp + (unsigned)(h << 3);
  unsigned boff1 = boff0 + ((unsigned)Kp << 4);   // +16 rows of Bt

  v8f acc0 = {};
  v8f acc1 = {};
  if (Cinit) {
#pragma unroll
    for (int v = 0; v < 8; ++v) {
      int m = m0 + v + (h << 3);              // C/D layout: VGPR v -> M = v + 8h
      acc0[v] = Cinit[(size_t)m * Np + n0 + lr];
      acc1[v] = Cinit[(size_t)m * Np + n0 + 16 + lr];
    }
  }
  for (int k0 = 0; k0 < Kp; k0 += 32, aoff += 32, boff0 += 32, boff1 += 32) {
    v16bf ah  = load_frag(Ahi,  aoff);
    v16bf al  = load_frag(Alo,  aoff);
    v16bf bh0 = load_frag(Bthi, boff0);
    v16bf bl0 = load_frag(Btlo, boff0);
    v16bf bh1 = load_frag(Bthi, boff1);
    v16bf bl1 = load_frag(Btlo, boff1);
    if (k0 + 64 < Kp) {                        // -> global_prefetch_b8, 2 steps ahead
      __builtin_prefetch(Ahi  + aoff  + 64, 0, 1);
      __builtin_prefetch(Bthi + boff0 + 64, 0, 1);
      __builtin_prefetch(Bthi + boff1 + 64, 0, 1);
    }
    // 2-term split: drop lo*lo (below f32 ulp). Interleave the two
    // independent accumulators to cover bf16 WMMA->WMMA RAW latency.
    acc0 = WMMA_BF16(al, bh0, acc0);
    acc1 = WMMA_BF16(al, bh1, acc1);
    acc0 = WMMA_BF16(ah, bl0, acc0);
    acc1 = WMMA_BF16(ah, bl1, acc1);
    acc0 = WMMA_BF16(ah, bh0, acc0);
    acc1 = WMMA_BF16(ah, bh1, acc1);
  }
#pragma unroll
  for (int v = 0; v < 8; ++v) {
    int m = m0 + v + (h << 3);
    if (m < Mreal) {
      size_t i0 = (size_t)m * Np + n0 + lr;
      size_t i1 = i0 + 16;
      if (Dout) { Dout[i0] = acc0[v]; Dout[i1] = acc1[v]; }
      if (Dhi)  { split2(acc0[v], Dhi + i0, Dlo + i0);
                  split2(acc1[v], Dhi + i1, Dlo + i1); }
    }
  }
}

// ------------------------------- launcher ----------------------------------

extern "C" void kernel_launch(void* const* d_in, const int* in_sizes, int n_in,
                              void* d_out, int out_size, void* d_ws, size_t ws_size,
                              hipStream_t stream) {
  (void)n_in; (void)out_size; (void)ws_size;
  const float* img    = (const float*)d_in[0];
  const float* attr   = (const float*)d_in[1];
  const int*   labels = (const int*)d_in[2];
  const int*   tpl    = (const int*)d_in[3];
  const float* attg   = (const float*)d_in[4];
  const float* cent   = (const float*)d_in[5];
  const float* eW     = (const float*)d_in[6];
  const float* eb     = (const float*)d_in[7];
  float* out = (float*)d_out;

  const int N  = in_sizes[2];                 // 100000
  const int C  = in_sizes[3];                 // 1000
  const int D  = in_sizes[1] / C;             // 512 (att dim)
  const int H  = in_sizes[4] / D;             // 512
  const int K  = in_sizes[5] / D;             // 10
  const int DI = in_sizes[0] / N;             // 2048
  const int CP = ((C + 31) / 32) * 32;        // 1024 padded classes
  const long long KD = (long long)K * D;      // 5120

  // workspace carve-out (256B aligned)
  char* ws = (char*)d_ws;
  unsigned long long off = 0;
  auto carve = [&](unsigned long long bytes) -> char* {
    off = (off + 255ull) & ~255ull;
    char* p = ws + off;
    off += bytes;
    return p;
  };
  float* sums    = (float*)carve((size_t)CP * DI * 4);
  float* counts  = (float*)carve((size_t)CP * 4);
  float* w       = (float*)carve((size_t)CP * K * 4);
  float* biasmix = (float*)carve((size_t)CP * D * 4);
  u16* Zhi    = (u16*)carve((size_t)CP * KD * 2);
  u16* Zlo    = (u16*)carve((size_t)CP * KD * 2);
  u16* W2thi  = (u16*)carve((size_t)D * KD * 2);
  u16* W2tlo  = (u16*)carve((size_t)D * KD * 2);
  u16* enchi  = (u16*)carve((size_t)CP * D * 2);
  u16* enclo  = (u16*)carve((size_t)CP * D * 2);
  u16* agThi  = (u16*)carve((size_t)H * D * 2);
  u16* agTlo  = (u16*)carve((size_t)H * D * 2);
  float* g    = (float*)carve((size_t)CP * H * 4);
  u16* gnhi   = (u16*)carve((size_t)CP * H * 2);
  u16* gnlo   = (u16*)carve((size_t)CP * H * 2);
  float* sim  = (float*)carve((size_t)CP * CP * 4);
  u16* athi   = (u16*)carve((size_t)CP * CP * 2);
  u16* atlo   = (u16*)carve((size_t)CP * CP * 2);
  u16* pThi   = (u16*)carve((size_t)DI * CP * 2);
  u16* pTlo   = (u16*)carve((size_t)DI * CP * 2);

  // 1) zero atomic accumulators (only buffers not fully overwritten later)
  k_zero<<<2048, 256, 0, stream>>>(sums, (long long)CP * DI);
  k_zero<<<8, 256, 0, stream>>>(counts, (long long)CP);

  // 2) segment sums + counts (HBM-bound streaming pass)
  k_segsum<<<N, 256, 0, stream>>>(img, labels, sums, counts, N, DI);

  // 3) mixture weights + bias mix
  k_mixw<<<(CP + 255) / 256, 256, 0, stream>>>(attr, cent, eb, w, biasmix, C, CP, K, D);

  // 4) operand prep (bf16 hi/lo splits, B pre-transposed)
  k_buildZ<<<4096, 256, 0, stream>>>(attr, w, Zhi, Zlo, C, CP, K, D);
  k_buildW2t<<<2048, 256, 0, stream>>>(eW, W2thi, W2tlo, K, D);
  k_buildAttgT<<<256, 256, 0, stream>>>(attg, agThi, agTlo, D, H);

  auto gemm_blocks = [](int Mp, int Np) { return ((Mp >> 4) * (Np >> 5) + 7) / 8; };

  // 5) enc = Z(CPxKD) @ W2(KDxD) + biasmix   -> bf16 hi/lo
  k_gemm<<<gemm_blocks(CP, D), 256, 0, stream>>>(
      Zhi, Zlo, W2thi, W2tlo, biasmix, nullptr, enchi, enclo, CP, D, (int)KD, CP);

  // 6) g = enc(CPxD) @ att_g(DxH)            -> f32
  k_gemm<<<gemm_blocks(CP, H), 256, 0, stream>>>(
      enchi, enclo, agThi, agTlo, nullptr, g, nullptr, nullptr, CP, H, D, CP);

  // 7) row-normalize -> gn bf16 hi/lo
  k_rownorm<<<CP, 256, 0, stream>>>(g, gnhi, gnlo, H);

  // 8) sim = gn @ gn^T  (B^T == gn, no transpose needed)
  k_gemm<<<gemm_blocks(CP, CP), 256, 0, stream>>>(
      gnhi, gnlo, gnhi, gnlo, nullptr, sim, nullptr, nullptr, CP, CP, H, CP);

  // 9) threshold + temperature + row softmax -> attn bf16 hi/lo
  k_attn_softmax<<<CP, 256, 0, stream>>>(sim, athi, atlo, C, CP);

  // 10) protos (gathered means) pre-transposed -> bf16 hi/lo
  k_protoT<<<2048, 256, 0, stream>>>(sums, counts, tpl, pThi, pTlo, C, CP, DI);

  // 11) out = attn(CPxCP) @ protos(CPxDI), store only first C rows
  k_gemm<<<gemm_blocks(CP, DI), 256, 0, stream>>>(
      athi, atlo, pThi, pTlo, nullptr, out, nullptr, nullptr, CP, DI, CP, C);
}
